// CNNBackbone_83468394430599
// MI455X (gfx1250) — matrix operand order repaired and checked by reading, not compiled, not run on hardware
//
#include <hip/hip_runtime.h>
#include <hip/hip_bf16.h>

// ---------------------------------------------------------------------------
// CNN backbone for MI455X (gfx1250), wave32 + WMMA f16 (f32 accumulate).
//   B=32, S=2048, D=128, C=256, K=5
// Pipeline (all on `stream`):
//   prep_x      : f32 -> f16 activations
//   prep_small  : repack conv weights per-tap [tap][Cout][Cin] (f16),
//                 fold BN (scale/shift), copy final bias
//   conv_bn_relu<128> : conv1 + BN + ReLU   (X16 -> H1, f16)
//   conv_bn_relu<256> : conv2 + BN + ReLU   (H1  -> H2, f16)
//   head_pool   : 1x1 conv + bias + masked partial pooling (deterministic)
//   finalize    : reduce partials, divide by length
// Intermediates (f16, ~83 MB workspace) stay resident in the 192MB L2.
// GEMM loops are branch-free and pipeline-shaped with sched_group_barrier
// ([2x VMEM_READ][1x WMMA]) so each WMMA waits with loads still in flight.
// ---------------------------------------------------------------------------

#define BB 32
#define SS 2048
#define DDIM 128
#define CC 256
#define STILE 64          // S rows per block
#define NTILES (SS / STILE)

typedef __attribute__((ext_vector_type(16))) _Float16 v16h;
typedef __attribute__((ext_vector_type(8)))  float    v8f;
typedef __attribute__((ext_vector_type(4)))  _Float16 v4h;

union FragH { v16h v; uint4 q[2]; };

static __device__ __forceinline__ float relu_f(float x) { return x > 0.f ? x : 0.f; }

// Scheduler shaping: request groups of 2 VMEM reads / 1 WMMA so fragment
// loads interleave with matrix ops instead of collapsing to load->wait0->wmma.
static __device__ __forceinline__ void sched_vmem2() {
#if __has_builtin(__builtin_amdgcn_sched_group_barrier)
    __builtin_amdgcn_sched_group_barrier(0x020, 2, 0);   // 2x VMEM read
#endif
}
static __device__ __forceinline__ void sched_wmma1() {
#if __has_builtin(__builtin_amdgcn_sched_group_barrier)
    __builtin_amdgcn_sched_group_barrier(0x008, 1, 0);   // 1x MFMA/WMMA
#endif
}

// ---------------------------------------------------------------------------
// Fused conv1d(K=5, pad=2) + BN + ReLU as 5 shifted GEMMs.
//   A : activations [B][S][CIN]  f16 (row-major, rows 16B-aligned)
//   Wt: weights     [5][CC][CIN] f16 (tap-major; row = Cout, Cin contiguous)
//   O : output      [B][S][CC]   f16
// Block: 256 threads = 8 waves. Wave w: M-tile m = w&3 (16 S rows),
// N-tiles nb0 = (w>>2)*8 .. +7 (8 x 16 channels). 64x256 output per block.
// ---------------------------------------------------------------------------
template <int CIN>
__global__ __launch_bounds__(256) void conv_bn_relu(
    const _Float16* __restrict__ A, const _Float16* __restrict__ Wt,
    const float* __restrict__ scale, const float* __restrict__ shift,
    _Float16* __restrict__ O)
{
    const int tid  = threadIdx.x;
    const int wave = tid >> 5, lane = tid & 31;
    const int r = lane & 15, hh = lane >> 4;          // row-in-tile, K-half
    const int m = wave & 3, nb0 = (wave >> 2) * 8;
    const int b = blockIdx.y, s0 = blockIdx.x * STILE;
    const _Float16* Ab = A + (size_t)b * SS * CIN;
    const int mrow = s0 + m * 16 + r;                 // this lane's M row

    constexpr int NKB = CIN / 32;                     // K-blocks per tap
    constexpr int TOT = 5 * NKB;                      // total K steps

    v8f acc[8];
#pragma unroll
    for (int j = 0; j < 8; ++j)
        acc[j] = v8f{0.f,0.f,0.f,0.f,0.f,0.f,0.f,0.f};

    // A fragment (16x32 f16): q0 @ k0+h*8, q1 @ k0+16+h*8 (ISA A layout).
    // Branch-free halo: clamp the row, always load, zero via VALU mask.
    auto loadA = [&](int t) -> FragH {
        const int tap = t / NKB, kb = t % NKB;
        const int srow = mrow + tap - 2;
        const bool valid = (srow >= 0) && (srow < SS);
        int srow_c = srow < 0 ? 0 : srow;
        srow_c = srow_c > (SS - 1) ? (SS - 1) : srow_c;
        const uint4* pa = reinterpret_cast<const uint4*>(
            Ab + (size_t)srow_c * CIN + kb * 32 + hh * 8);
        FragH f;
        f.q[0] = pa[0];
        f.q[1] = pa[2];
        const unsigned msk = valid ? 0xFFFFFFFFu : 0u;
        f.q[0].x &= msk; f.q[0].y &= msk; f.q[0].z &= msk; f.q[0].w &= msk;
        f.q[1].x &= msk; f.q[1].y &= msk; f.q[1].z &= msk; f.q[1].w &= msk;
        return f;
    };
    // B fragment (32x16 f16): lane = N row, q0 @ k0+h*8, q1 @ k0+16+h*8 (matches A).
    auto loadB = [&](int t, int j) -> FragH {
        const int tap = t / NKB, kb = t % NKB;
        const uint4* pb = reinterpret_cast<const uint4*>(
            Wt + (size_t)tap * CC * CIN
               + (size_t)((nb0 + j) * 16 + r) * CIN + kb * 32 + hh * 8);
        FragH f;
        f.q[0] = pb[0];
        f.q[1] = pb[2];
        return f;
    };

    // Software pipeline: fragments consumed at step t / slot j were issued one
    // step earlier; branch-free body keeps it one scheduling region.
    FragH af = loadA(0);
    FragH bf = loadB(0, 0);
    for (int t = 0; t < TOT; ++t) {
        const int tn = (t + 1 < TOT) ? (t + 1) : 0;   // uniform wrap, no branch
        FragH af_n = loadA(tn);
        sched_vmem2();                                // group: A_next pair
#pragma unroll
        for (int j = 0; j < 8; ++j) {
            FragH bf_n = (j < 7) ? loadB(t, j + 1) : loadB(tn, 0);
            acc[j] = __builtin_amdgcn_wmma_f32_16x16x32_f16(
                false, af.v, false, bf.v, (short)0, acc[j], false, false);
            bf = bf_n;
            sched_vmem2();                            // group: B_next pair
            sched_wmma1();                            // group: this WMMA
        }
        af = af_n;
    }

    // Epilogue: BN scale/shift + ReLU, cast f16, store.
    // C/D layout: VGPR rr holds M=rr (lanes 0-15) / M=rr+8 (lanes 16-31);
    // N = lane&15 -> channel is uniform per lane across all 8 VGPRs.
#pragma unroll
    for (int j = 0; j < 8; ++j) {
        const int c = (nb0 + j) * 16 + r;
        const float sc = scale[c], sh = shift[c];
        _Float16* orow = O + ((size_t)b * SS + s0 + m * 16 + hh * 8) * CC + c;
#pragma unroll
        for (int rr = 0; rr < 8; ++rr) {
            const float v = fmaf(acc[j][rr], sc, sh);
            orow[(size_t)rr * CC] = (_Float16)relu_f(v);
        }
    }
}

// ---------------------------------------------------------------------------
// 1x1 conv (C=256 -> D=128) + bias + masked in-tile pooling.
// partial[b][tile][m][dd] = sum over this 16-row M-tile of masked conv output.
// ---------------------------------------------------------------------------
__global__ __launch_bounds__(256) void head_pool(
    const _Float16* __restrict__ H2, const _Float16* __restrict__ Wtf,
    const float* __restrict__ bfc, const int* __restrict__ spi,
    float* __restrict__ partial)
{
    const int tid  = threadIdx.x;
    const int wave = tid >> 5, lane = tid & 31;
    const int r = lane & 15, hh = lane >> 4;
    const int m = wave & 3, ng = wave >> 2;           // n = ng*4 + j, j<4
    const int b = blockIdx.y, s0 = blockIdx.x * STILE;
    const _Float16* Ab = H2 + (size_t)b * SS * CC;
    const int mrow = s0 + m * 16 + r;

    v8f acc[4];
#pragma unroll
    for (int j = 0; j < 4; ++j)
        acc[j] = v8f{0.f,0.f,0.f,0.f,0.f,0.f,0.f,0.f};

    auto loadA = [&](int kb) -> FragH {
        const uint4* pa = reinterpret_cast<const uint4*>(
            Ab + (size_t)mrow * CC + (kb & 7) * 32 + hh * 8);
        FragH f;
        f.q[0] = pa[0];
        f.q[1] = pa[2];
        return f;
    };
    auto loadB = [&](int kb, int j) -> FragH {
        const uint4* pb = reinterpret_cast<const uint4*>(
            Wtf + (size_t)((ng * 4 + j) * 16 + r) * CC + (kb & 7) * 32 + hh * 8);
        FragH f;
        f.q[0] = pb[0];
        f.q[1] = pb[2];
        return f;
    };

    FragH af = loadA(0);
    FragH bf = loadB(0, 0);
#pragma unroll
    for (int kb = 0; kb < 8; ++kb) {
        FragH af_n = loadA(kb + 1);                   // kb==7 wraps (dummy)
        sched_vmem2();
#pragma unroll
        for (int j = 0; j < 4; ++j) {
            FragH bf_n = (j < 3) ? loadB(kb, j + 1) : loadB(kb + 1, 0);
            acc[j] = __builtin_amdgcn_wmma_f32_16x16x32_f16(
                false, af.v, false, bf.v, (short)0, acc[j], false, false);
            bf = bf_n;
            sched_vmem2();
            sched_wmma1();
        }
        af = af_n;
    }

    int len = spi[b];
    if (len == -1) len = SS;

#pragma unroll
    for (int j = 0; j < 4; ++j) {
        const int dd = (ng * 4 + j) * 16 + r;
        const float bias = bfc[dd];
        float lsum = 0.f;
#pragma unroll
        for (int rr = 0; rr < 8; ++rr) {
            const int s = s0 + m * 16 + rr + hh * 8;
            if (s < len) lsum += acc[j][rr] + bias;
        }
        // lanes l and l+16 hold the same dd (N = lane&15) for disjoint rows
        lsum += __shfl_xor(lsum, 16, 32);
        if (hh == 0)
            partial[((size_t)(b * NTILES + blockIdx.x) * 4 + m) * DDIM + dd] = lsum;
    }
}

__global__ __launch_bounds__(256) void finalize_pool(
    const float* __restrict__ partial, const int* __restrict__ spi,
    float* __restrict__ out)
{
    const int tid = blockIdx.x * blockDim.x + threadIdx.x;
    if (tid >= BB * DDIM) return;
    const int b = tid >> 7, dd = tid & 127;
    int len = spi[b];
    if (len == -1) len = SS;
    float s = 0.f;
    for (int u = 0; u < NTILES * 4; ++u)
        s += partial[((size_t)b * NTILES * 4 + u) * DDIM + dd];
    out[tid] = (len > 0) ? (s / (float)len) : 0.f;
}

// ---------------------------------------------------------------------------
// Prep kernels
// ---------------------------------------------------------------------------
__global__ __launch_bounds__(256) void prep_x(
    const float* __restrict__ x, _Float16* __restrict__ X16, int n4)
{
    const int i = blockIdx.x * 256 + threadIdx.x;
    if (i >= n4) return;
    const float4 f = reinterpret_cast<const float4*>(x)[i];
    v4h h = {(_Float16)f.x, (_Float16)f.y, (_Float16)f.z, (_Float16)f.w};
    reinterpret_cast<v4h*>(X16)[i] = h;
}

__global__ __launch_bounds__(256) void prep_small(
    const float* __restrict__ W1, const float* __restrict__ W2,
    const float* __restrict__ Wf,
    const float* __restrict__ b1, const float* __restrict__ g1,
    const float* __restrict__ be1, const float* __restrict__ m1,
    const float* __restrict__ v1,
    const float* __restrict__ b2, const float* __restrict__ g2,
    const float* __restrict__ be2, const float* __restrict__ m2,
    const float* __restrict__ v2, const float* __restrict__ bfv,
    _Float16* __restrict__ Wt1, _Float16* __restrict__ Wt2,
    _Float16* __restrict__ Wtf,
    float* __restrict__ scale1, float* __restrict__ shift1,
    float* __restrict__ scale2, float* __restrict__ shift2,
    float* __restrict__ bfc)
{
    const int idx = blockIdx.x * 256 + threadIdx.x;
    // Wt1[tap][c][d] = W1[c][d][tap]   (W1: [C, D, 5] row-major)
    if (idx < 5 * CC * DDIM) {
        const int tap = idx >> 15, c = (idx >> 7) & 255, d = idx & 127;
        Wt1[idx] = (_Float16)W1[(c * DDIM + d) * 5 + tap];
    }
    // Wt2[tap][c][c2] = W2[c][c2][tap] (W2: [C, C, 5] row-major)
    if (idx < 5 * CC * CC) {
        const int tap = idx >> 16, c = (idx >> 8) & 255, c2 = idx & 255;
        Wt2[idx] = (_Float16)W2[(c * CC + c2) * 5 + tap];
    }
    // Wtf[dd][c] = Wf[dd][c][0]        (Wf: [D, C, 1] row-major, already B-layout)
    if (idx < DDIM * CC) Wtf[idx] = (_Float16)Wf[idx];
    if (idx < CC) {
        const float s1 = g1[idx] * rsqrtf(v1[idx] + 1e-5f);
        scale1[idx] = s1;
        shift1[idx] = (b1[idx] - m1[idx]) * s1 + be1[idx];
        const float s2 = g2[idx] * rsqrtf(v2[idx] + 1e-5f);
        scale2[idx] = s2;
        shift2[idx] = (b2[idx] - m2[idx]) * s2 + be2[idx];
    }
    if (idx < DDIM) bfc[idx] = bfv[idx];
}

// ---------------------------------------------------------------------------
extern "C" void kernel_launch(void* const* d_in, const int* in_sizes, int n_in,
                              void* d_out, int out_size, void* d_ws, size_t ws_size,
                              hipStream_t stream)
{
    (void)in_sizes; (void)n_in; (void)out_size; (void)ws_size;

    const float* x   = (const float*)d_in[0];
    const int*   spi = (const int*)  d_in[1];
    const float* W1  = (const float*)d_in[2];
    const float* b1  = (const float*)d_in[3];
    const float* g1  = (const float*)d_in[4];
    const float* be1 = (const float*)d_in[5];
    const float* m1  = (const float*)d_in[6];
    const float* v1  = (const float*)d_in[7];
    const float* W2  = (const float*)d_in[8];
    const float* b2  = (const float*)d_in[9];
    const float* g2  = (const float*)d_in[10];
    const float* be2 = (const float*)d_in[11];
    const float* m2  = (const float*)d_in[12];
    const float* v2  = (const float*)d_in[13];
    const float* Wf  = (const float*)d_in[14];
    const float* bfv = (const float*)d_in[15];

    char*  ws  = (char*)d_ws;
    size_t off = 0;
    auto alloc = [&](size_t bytes) -> void* {
        void* p = ws + off;
        off += (bytes + 255) & ~(size_t)255;
        return p;
    };
    _Float16* X16 = (_Float16*)alloc((size_t)BB * SS * DDIM * 2);  // 16 MB
    _Float16* H1  = (_Float16*)alloc((size_t)BB * SS * CC * 2);    // 32 MB
    _Float16* H2  = (_Float16*)alloc((size_t)BB * SS * CC * 2);    // 32 MB
    _Float16* Wt1 = (_Float16*)alloc((size_t)5 * CC * DDIM * 2);
    _Float16* Wt2 = (_Float16*)alloc((size_t)5 * CC * CC * 2);
    _Float16* Wtf = (_Float16*)alloc((size_t)DDIM * CC * 2);
    float* scale1 = (float*)alloc(CC * 4);
    float* shift1 = (float*)alloc(CC * 4);
    float* scale2 = (float*)alloc(CC * 4);
    float* shift2 = (float*)alloc(CC * 4);
    float* bfc    = (float*)alloc(DDIM * 4);
    float* partial = (float*)alloc((size_t)BB * NTILES * 4 * DDIM * 4); // 2 MB

    const int n4 = BB * SS * DDIM / 4;
    prep_x<<<(n4 + 255) / 256, 256, 0, stream>>>(x, X16, n4);
    prep_small<<<(5 * CC * CC + 255) / 256, 256, 0, stream>>>(
        W1, W2, Wf, b1, g1, be1, m1, v1, b2, g2, be2, m2, v2, bfv,
        Wt1, Wt2, Wtf, scale1, shift1, scale2, shift2, bfc);

    dim3 grid(NTILES, BB);
    conv_bn_relu<DDIM><<<grid, 256, 0, stream>>>(X16, Wt1, scale1, shift1, H1);
    conv_bn_relu<CC ><<<grid, 256, 0, stream>>>(H1,  Wt2, scale2, shift2, H2);
    head_pool<<<grid, 256, 0, stream>>>(H2, Wtf, bfc, spi, partial);
    finalize_pool<<<(BB * DDIM + 255) / 256, 256, 0, stream>>>(
        partial, spi, (float*)d_out);
}
